// GroupedQueryAttention_45071386804276
// MI455X (gfx1250) — compile-verified, matrix operand order checked by reading, not run on hardware
//
#include <hip/hip_runtime.h>

typedef __attribute__((ext_vector_type(16))) _Float16 v16h;
typedef __attribute__((ext_vector_type(8)))  _Float16 v8h;
typedef __attribute__((ext_vector_type(2)))  _Float16 v2h;
typedef __attribute__((ext_vector_type(8)))  float    v8f;

#define HQ    16
#define HKV   4
#define DDIM  64
#define QSCALE 0.125f   // 1/sqrt(64)

// LDS padded strides (halves): multiples of 8 (16B alignment for b128
// vector access) and chosen so 16-lane row accesses hit distinct banks.
#define SK_STR  72     // 144B rows: 36-dword stride, 16 lanes -> 16 banks
#define SVT_STR 40     // 80B rows: 20-dword stride, conflict-free
#define SP_STR  40

__device__ __forceinline__ v8f wmma_f16(v16h a, v16h b, v8f c) {
  return __builtin_amdgcn_wmma_f32_16x16x32_f16(
      false, a, false, b, (short)0, c, false, false);
}

// Two contiguous 8-half LDS chunks -> one v16h fragment.
// (ISA 7.12.2: element i holds K = (i/8)*16 + half*8 + (i%8), so
// elements 0-7 and 8-15 are each contiguous runs of 8.)
__device__ __forceinline__ v16h frag_ld(const _Float16* lo, const _Float16* hi) {
  v8h a = *(const v8h*)lo;
  v8h b = *(const v8h*)hi;
  return __builtin_shufflevector(a, b, 0,1,2,3,4,5,6,7,8,9,10,11,12,13,14,15);
}

__device__ __forceinline__ v8h cvt8(float4 a, float4 b, float s) {
  v8h h;
  h[0]=(_Float16)(a.x*s); h[1]=(_Float16)(a.y*s);
  h[2]=(_Float16)(a.z*s); h[3]=(_Float16)(a.w*s);
  h[4]=(_Float16)(b.x*s); h[5]=(_Float16)(b.y*s);
  h[6]=(_Float16)(b.z*s); h[7]=(_Float16)(b.w*s);
  return h;
}

// Invariant (assumed by kernel_launch, not checked): S % 32 == 0.  Staged rows are
// t..t+31 with t a multiple of 32 and t < kend <= S, hence t <= S-32 and
// every staged row is in bounds -> NO per-element bounds checks needed.
__global__ __launch_bounds__(128)
void gqa_flash_fwd(const float* __restrict__ Q, const float* __restrict__ K,
                   const float* __restrict__ V, float* __restrict__ O, int S)
{
  __shared__ __align__(16) _Float16 sK [32][SK_STR];    // [key][chan]
  __shared__ __align__(16) _Float16 sVT[64][SVT_STR];   // [chan][key] (transposed)
  __shared__ __align__(16) _Float16 sP [4][16][SP_STR]; // per-wave [row][key]

  const int tid  = threadIdx.x;
  const int wave = tid >> 5;
  const int lane = tid & 31;
  const int half = lane >> 4;
  const int l16  = lane & 15;
  const int h8   = half << 3;

  const int q0  = blockIdx.x << 4;
  const int kvh = blockIdx.y;
  const int b   = blockIdx.z;
  const int qh  = (kvh << 2) + wave;

  const float* Qb = Q + (size_t)b * S * (HQ  * DDIM) + (size_t)qh  * DDIM;
  const float* Kb = K + (size_t)b * S * (HKV * DDIM) + (size_t)kvh * DDIM;
  const float* Vb = V + (size_t)b * S * (HKV * DDIM) + (size_t)kvh * DDIM;
  float*       Ob = O + ((size_t)b * HQ + qh) * S * DDIM;

  // ---- Q tile (16x64) -> two A fragments, pre-scaled, vector loads ----
  v16h aQ0, aQ1;
  {
    const float* qr = Qb + (size_t)(q0 + l16) * (HQ * DDIM);
    v8h q0lo = cvt8(*(const float4*)(qr + h8),      *(const float4*)(qr + h8 + 4),      QSCALE);
    v8h q0hi = cvt8(*(const float4*)(qr + 16 + h8), *(const float4*)(qr + 16 + h8 + 4), QSCALE);
    v8h q1lo = cvt8(*(const float4*)(qr + 32 + h8), *(const float4*)(qr + 32 + h8 + 4), QSCALE);
    v8h q1hi = cvt8(*(const float4*)(qr + 48 + h8), *(const float4*)(qr + 48 + h8 + 4), QSCALE);
    aQ0 = __builtin_shufflevector(q0lo, q0hi, 0,1,2,3,4,5,6,7,8,9,10,11,12,13,14,15);
    aQ1 = __builtin_shufflevector(q1lo, q1hi, 0,1,2,3,4,5,6,7,8,9,10,11,12,13,14,15);
  }

  // all-ones B fragment: row-sum of P via one WMMA (replicated across lanes)
  v16h vones;
#pragma unroll
  for (int i = 0; i < 16; ++i) vones[i] = (_Float16)1.0f;

  v8f acc[4] = {};
  float mrow[8], lrow[8];
#pragma unroll
  for (int j = 0; j < 8; ++j) { mrow[j] = -__builtin_inff(); lrow[j] = 0.f; }

  // staging thread mappings (loop-invariant)
  const int kKey = tid >> 3;                 // K stage: key 0..15 (+16)
  const int kC8  = (tid & 7) << 3;           // K stage: 8-chan block
  const int vCh  = tid & 63;                 // V stage: channel
  const int vKb  = (tid >> 6) << 4;          // V stage: key block 0/16

  const int kend = q0 + 16;
  for (int t = 0; t < kend; t += 32) {
    const float* Kt = Kb + (size_t)t * (HKV * DDIM);
    const float* Vt = Vb + (size_t)t * (HKV * DDIM);

    // ---- stage K (row-major), branchless, b128 in / b128 out ----
#pragma unroll
    for (int it = 0; it < 2; ++it) {
      int key = kKey + (it << 4);
      const float* src = Kt + (size_t)key * (HKV * DDIM) + kC8;
      float4 f0 = *(const float4*)(src);
      float4 f1 = *(const float4*)(src + 4);
      *(v8h*)&sK[key][kC8] = cvt8(f0, f1, 1.0f);
    }
    // ---- stage V transposed: one chan x 16 keys/thread, b128 stores ----
    {
      const float* src = Vt + (size_t)vKb * (HKV * DDIM) + vCh;
      _Float16 tmp[16];
#pragma unroll
      for (int i = 0; i < 16; ++i)           // lanes sweep chans: coalesced
        tmp[i] = (_Float16)src[(size_t)i * (HKV * DDIM)];
      *(v8h*)&sVT[vCh][vKb]     = *(const v8h*)&tmp[0];
      *(v8h*)&sVT[vCh][vKb + 8] = *(const v8h*)&tmp[8];
    }
    __syncthreads();

    if (t + 32 < kend) {
      int pr = tid >> 2, pc = (tid & 3) << 4;
      __builtin_prefetch(Kt + (size_t)(32 + pr) * (HKV * DDIM) + pc, 0, 1);
      __builtin_prefetch(Vt + (size_t)(32 + pr) * (HKV * DDIM) + pc, 0, 1);
    }

    // ---- scores: S = Q * K^T, two 16x16 tiles (keys t.., t+16..) ----
    v16h b0lo = frag_ld(&sK[l16][h8],           &sK[l16][16 + h8]);
    v16h b0hi = frag_ld(&sK[l16][32 + h8],      &sK[l16][48 + h8]);
    v16h b1lo = frag_ld(&sK[l16 + 16][h8],      &sK[l16 + 16][16 + h8]);
    v16h b1hi = frag_ld(&sK[l16 + 16][32 + h8], &sK[l16 + 16][48 + h8]);
    v8f s0 = {}, s1 = {};
    s0 = wmma_f16(aQ0, b0lo, s0);
    s0 = wmma_f16(aQ1, b0hi, s0);
    s1 = wmma_f16(aQ0, b1lo, s1);
    s1 = wmma_f16(aQ1, b1hi, s1);

    // ---- causal mask on diagonal-overlapping steps (uniform branch) ----
    if (t + 31 > q0) {
#pragma unroll
      for (int j = 0; j < 8; ++j) {
        int row = q0 + j + h8;               // D layout: M = j + 8*half
        if (t + l16 > row)      s0[j] = -__builtin_inff();
        if (t + 16 + l16 > row) s1[j] = -__builtin_inff();
      }
    }

    // ---- row max: pack 2 rows per f16x2 shuffle (16 bpermutes/step) ----
    float sc[8];
#pragma unroll
    for (int jp = 0; jp < 4; ++jp) {
      int j0 = jp << 1, j1 = j0 + 1;
      v2h pk;
      pk[0] = (_Float16)fmaxf(s0[j0], s1[j0]);
      pk[1] = (_Float16)fmaxf(s0[j1], s1[j1]);
#pragma unroll
      for (int m = 1; m <= 8; m <<= 1) {
        v2h o = __builtin_bit_cast(v2h, __shfl_xor(__builtin_bit_cast(int, pk), m, 32));
        pk[0] = (pk[0] > o[0]) ? pk[0] : o[0];
        pk[1] = (pk[1] > o[1]) ? pk[1] : o[1];
      }
      float rm[2] = { (float)pk[0], (float)pk[1] };
#pragma unroll
      for (int u = 0; u < 2; ++u) {
        int j = (u == 0) ? j0 : j1;
        float mn = fmaxf(mrow[j], rm[u]);
        float s  = __expf(mrow[j] - mn);     // exp(-inf)=0 on first touch
        float p0 = __expf(s0[j] - mn);
        float p1 = __expf(s1[j] - mn);
        mrow[j] = mn; sc[j] = s;
        acc[0][j] *= s; acc[1][j] *= s; acc[2][j] *= s; acc[3][j] *= s;
        int prow = j + h8;
        sP[wave][prow][l16]      = (_Float16)p0; // key = t + l16
        sP[wave][prow][l16 + 16] = (_Float16)p1; // key = t + 16 + l16
      }
    }

    // in-wave D->A transpose through LDS (same-wave LDS ops are in order)
    __builtin_amdgcn_wave_barrier();
    asm volatile("s_wait_dscnt 0" ::: "memory");

    v16h aP = frag_ld(&sP[wave][l16][h8], &sP[wave][l16][16 + h8]);

    // row-sum of P via WMMA against ones: rsum[j] = sum_k P[row][k]
    v8f z = {};
    v8f rsum = wmma_f16(aP, vones, z);
#pragma unroll
    for (int j = 0; j < 8; ++j) lrow[j] = lrow[j] * sc[j] + rsum[j];

    // ---- O += P(16x32) * V(32x64): B frags from transposed sVT ----
#pragma unroll
    for (int n = 0; n < 4; ++n) {
      const _Float16* vr = &sVT[(n << 4) + l16][0];
      v16h bv = frag_ld(vr + h8, vr + 16 + h8);
      acc[n] = wmma_f16(aP, bv, acc[n]);
    }
    __syncthreads();
  }

  // ---- epilogue: normalize and store fp32 ----
#pragma unroll
  for (int j = 0; j < 8; ++j) {
    int row = q0 + j + h8;
    float inv = 1.0f / lrow[j];
#pragma unroll
    for (int n = 0; n < 4; ++n)
      Ob[(size_t)row * DDIM + (n << 4) + l16] = acc[n][j] * inv;
  }
}

extern "C" void kernel_launch(void* const* d_in, const int* in_sizes, int n_in,
                              void* d_out, int out_size, void* d_ws, size_t ws_size,
                              hipStream_t stream) {
  (void)n_in; (void)out_size; (void)d_ws; (void)ws_size;
  const float* Q = (const float*)d_in[0];
  const float* K = (const float*)d_in[1];
  const float* V = (const float*)d_in[2];
  float* O = (float*)d_out;

  long long kE = in_sizes[1];                    // B*S*HKV*64
  long long mE = in_sizes[3];                    // B*S*S
  int S = (int)((mE / kE) * (HKV * DDIM));
  int B = (int)(kE / ((long long)S * HKV * DDIM));
  // Kernel requires S % 32 == 0 (true for this workload: S = 2048).

  dim3 grid(S / 16, HKV, B);
  gqa_flash_fwd<<<grid, 128, 0, stream>>>(Q, K, V, O, S);
}
